// EncoderStack_73014444032756
// MI455X (gfx1250) — compile-verified, matrix operand order checked by reading, not run on hardware
//
#include <hip/hip_runtime.h>

// ---------------------------------------------------------------------------
// CDNA5 (gfx1250) transformer encoder stack.
// wave32, WMMA 16x16x32 f16 -> f32, vectorized/pipelined LDS staging.
// ---------------------------------------------------------------------------

typedef __attribute__((ext_vector_type(16))) _Float16 v16h;
typedef __attribute__((ext_vector_type(8)))  _Float16 v8h;
typedef __attribute__((ext_vector_type(4)))  _Float16 f16x4;
typedef __attribute__((ext_vector_type(2)))  _Float16 f16x2;
typedef __attribute__((ext_vector_type(8)))  float    v8f;
typedef __attribute__((ext_vector_type(4)))  float    f32x4;

union Frag16 { v16h v; v8h h[2]; _Float16 e[16]; };

// Fragment for V_WMMA_F32_16X16X32_F16 from a row-major f32 row (16B loads).
// 16-bit A/B layout (ISA 7.12.2): hi = lane>>4 selects the K-half:
//   element e<8  -> K = koff + hi*8 + e
//   element e>=8 -> K = koff + 16 + hi*8 + (e-8)
__device__ inline v16h frag_from_f32row(const float* __restrict__ row, int hi, int koff) {
  const f32x4* p = (const f32x4*)(row + koff + hi * 8);
  const f32x4* q = (const f32x4*)(row + koff + 16 + hi * 8);
  f32x4 a0 = p[0], a1 = p[1], a2 = q[0], a3 = q[1];
  Frag16 u;
#pragma unroll
  for (int j = 0; j < 4; ++j) {
    u.e[j]      = (_Float16)a0[j];
    u.e[4 + j]  = (_Float16)a1[j];
    u.e[8 + j]  = (_Float16)a2[j];
    u.e[12 + j] = (_Float16)a3[j];
  }
  return u.v;
}

// ---------------------------------------------------------------------------
// Generic batched GEMM: C[b] = act(A[b] * B[b] + bias[b])
// A: MxK row-major (lda), B: KxN row-major (ldb), C: MxN row-major (ldc).
// 256 threads = 8 waves; block tile 128 x (16*NSUB); K step 32.
// Register-pipelined global->LDS staging (f32 -> f16), vectorized both ways.
// NSUB = 4 (64-wide) for small-N GEMMs, NSUB = 8 (128-wide) for the big ones.
// ---------------------------------------------------------------------------
#define LDSP 40  // padded LDS row stride (f16 elems) to spread banks

template <int NSUB>
__global__ __launch_bounds__(256)
void gemm_f16_kernel(const float* __restrict__ A, int lda, long long sA,
                     const float* __restrict__ B, int ldb, long long sB,
                     const float* __restrict__ bias, long long sBias,
                     float* __restrict__ C, int ldc, long long sC,
                     int M, int N, int K, int relu) {
  constexpr int NCOLS = NSUB * 16;       // 64 or 128
  constexpr int NGRP  = NCOLS / 64;      // 64-col groups per thread in staging

  __shared__ __attribute__((aligned(16))) _Float16 Al[128 * LDSP];
  __shared__ __attribute__((aligned(16))) _Float16 Bl[NCOLS * LDSP];

  int bz = blockIdx.z;
  A += (long long)bz * sA;
  B += (long long)bz * sB;
  C += (long long)bz * sC;
  if (bias) bias += (long long)bz * sBias;

  int bm = blockIdx.y * 128;
  int bn = blockIdx.x * NCOLS;
  int t    = threadIdx.x;   // 0..255
  int wave = t >> 5;        // 0..7 -> 16-row strip of the 128-row tile
  int lane = t & 31;
  int hi   = lane >> 4;
  int lo   = lane & 15;

  // Staging geometry.
  // A tile 128x32 = 1024 float4; thread does 4 (row r, 16B chunk c4).
  // B tile 32xNCOLS: per 64-col group, thread owns rows {kk2,kk2+1} x cols n4..n4+3.
  int kk2 = (t >> 4) * 2;
  int n4  = (t & 15) * 4;

  f32x4 aR[4];
  f32x4 bR[NGRP][2];

  auto loadA = [&](int k0) {
#pragma unroll
    for (int j = 0; j < 4; ++j) {
      int i = t + 256 * j;
      int r = i >> 3, c4 = (i & 7) * 4;
      aR[j] = *(const f32x4*)&A[(long long)(bm + r) * lda + k0 + c4];
    }
  };
  auto loadB = [&](int k0) {
#pragma unroll
    for (int g = 0; g < NGRP; ++g)
#pragma unroll
      for (int x = 0; x < 2; ++x)
        bR[g][x] = *(const f32x4*)&B[(long long)(k0 + kk2 + x) * ldb + bn + g * 64 + n4];
  };
  auto storeA = [&]() {
#pragma unroll
    for (int j = 0; j < 4; ++j) {
      int i = t + 256 * j;
      int r = i >> 3, c4 = (i & 7) * 4;
      f16x4 hh;
#pragma unroll
      for (int x = 0; x < 4; ++x) hh[x] = (_Float16)aR[j][x];
      *(f16x4*)&Al[r * LDSP + c4] = hh;   // ds_store_b64
    }
  };
  auto storeB = [&]() {
#pragma unroll
    for (int g = 0; g < NGRP; ++g)
#pragma unroll
      for (int y = 0; y < 4; ++y) {      // transpose 2x4 micro-tile in regs
        f16x2 hh;
        hh[0] = (_Float16)bR[g][0][y];
        hh[1] = (_Float16)bR[g][1][y];
        *(f16x2*)&Bl[(g * 64 + n4 + y) * LDSP + kk2] = hh;  // ds_store_b32
      }
  };

  v8f acc[NSUB] = {};

  loadA(0);
  loadB(0);
  for (int k0 = 0; k0 < K; k0 += 32) {
    storeA();
    storeB();
    __syncthreads();

    // Issue next slab's global loads so they overlap the WMMAs below.
    if (k0 + 32 < K) { loadA(k0 + 32); loadB(k0 + 32); }
    if (k0 + 64 < K) {
      __builtin_prefetch(&A[(long long)(bm + (t >> 1)) * lda + (k0 + 64)], 0, 1);
      __builtin_prefetch(&B[(long long)(k0 + 64 + (t >> 3)) * ldb + bn + (t & 7) * (NCOLS / 8)], 0, 1);
    }

    Frag16 af;
    int arow = wave * 16 + lo;
    af.h[0] = *(const v8h*)&Al[arow * LDSP + hi * 8];
    af.h[1] = *(const v8h*)&Al[arow * LDSP + 16 + hi * 8];
#pragma unroll
    for (int tn = 0; tn < NSUB; ++tn) {
      Frag16 bf;
      int brow = tn * 16 + lo;
      bf.h[0] = *(const v8h*)&Bl[brow * LDSP + hi * 8];
      bf.h[1] = *(const v8h*)&Bl[brow * LDSP + 16 + hi * 8];
      acc[tn] = __builtin_amdgcn_wmma_f32_16x16x32_f16(false, af.v, false, bf.v,
                                                       (short)0, acc[tn], false, false);
    }
    __syncthreads();
  }

  // Epilogue. C 16x16 f32 layout: VGPR v, lane -> row = v + 8*hi, col = lo.
#pragma unroll
  for (int tn = 0; tn < NSUB; ++tn) {
    int col = bn + tn * 16 + lo;
    float bv = bias ? bias[col] : 0.0f;
#pragma unroll
    for (int v = 0; v < 8; ++v) {
      int row = bm + wave * 16 + hi * 8 + v;
      float x = acc[tn][v] + bv;
      if (relu) x = fmaxf(x, 0.0f);
      C[(long long)row * ldc + col] = x;
    }
  }
}

// ---------------------------------------------------------------------------
// Column logsumexp of qk = Q K^T / sqrt(dk) over the query axis.
// Q,K: [H, N, 64] f32. One wave per (head, 16-key strip); B fragments (K^T)
// are loop-invariant; online max/sum over N/16 query tiles.
// ---------------------------------------------------------------------------
__global__ __launch_bounds__(32)
void lse_kernel(const float* __restrict__ Q, const float* __restrict__ Km,
                float* __restrict__ lseOut, int Nn) {
  int h     = blockIdx.y;
  int mbase = blockIdx.x * 16;
  int lane  = threadIdx.x;
  int hi = lane >> 4, lo = lane & 15;

  const float* Qh = Q  + (long long)h * Nn * 64;
  const float* Kh = Km + (long long)h * Nn * 64;

  // B = K^T (64 x 16): column = key index m, K-dim = dk. B[k][m] = K[m][k].
  const float* krow = Kh + (long long)(mbase + lo) * 64;
  v16h b0 = frag_from_f32row(krow, hi, 0);
  v16h b1 = frag_from_f32row(krow, hi, 32);

  float mrun = -3.0e38f;
  float srun = 0.0f;

  for (int qt = 0; qt < Nn / 16; ++qt) {
    const float* qrow = Qh + (long long)(qt * 16 + lo) * 64;
    v16h a0 = frag_from_f32row(qrow, hi, 0);
    v16h a1 = frag_from_f32row(qrow, hi, 32);
    v8f accv = {};
    accv = __builtin_amdgcn_wmma_f32_16x16x32_f16(false, a0, false, b0, (short)0, accv, false, false);
    accv = __builtin_amdgcn_wmma_f32_16x16x32_f16(false, a1, false, b1, (short)0, accv, false, false);

    float x[8];
    float tmax = -3.0e38f;
#pragma unroll
    for (int v = 0; v < 8; ++v) { x[v] = accv[v] * 0.125f; tmax = fmaxf(tmax, x[v]); }
    float nm = fmaxf(mrun, tmax);
    float s = srun * __expf(mrun - nm);
#pragma unroll
    for (int v = 0; v < 8; ++v) s += __expf(x[v] - nm);
    mrun = nm; srun = s;
  }

  // Lanes lo and lo+16 hold the two row-halves of column (mbase+lo): combine.
  float mo = __shfl_xor(mrun, 16, 32);
  float so = __shfl_xor(srun, 16, 32);
  float nm = fmaxf(mrun, mo);
  float tot = srun * __expf(mrun - nm) + so * __expf(mo - nm);
  if (hi == 0) lseOut[(long long)h * Nn + mbase + lo] = nm + __logf(tot);
}

// ---------------------------------------------------------------------------
// Mout[h] = K[h]^T V[h] / sqrt(dk)  (64x64),  negb[h] = -(lse[h] . V[h])  (64)
// Deterministic (no float atomics): 1024 threads, 4 outputs each.
// ---------------------------------------------------------------------------
__global__ __launch_bounds__(1024)
void ktv_kernel(const float* __restrict__ Km, const float* __restrict__ V,
                const float* __restrict__ lse, float* __restrict__ Mout,
                float* __restrict__ negb, int Nn) {
  int h = blockIdx.x;
  const float* Kh = Km + (long long)h * Nn * 64;
  const float* Vh = V  + (long long)h * Nn * 64;

  for (int idx = threadIdx.x; idx < 64 * 64; idx += 1024) {
    int kk = idx >> 6, vv = idx & 63;
    float acc = 0.0f;
    for (int m = 0; m < Nn; ++m)
      acc += Kh[(long long)m * 64 + kk] * Vh[(long long)m * 64 + vv];
    Mout[(long long)h * 4096 + idx] = acc * 0.125f;
  }
  if (threadIdx.x < 64) {
    int vv = threadIdx.x;
    float acc = 0.0f;
    for (int m = 0; m < Nn; ++m)
      acc += lse[(long long)h * Nn + m] * Vh[(long long)m * 64 + vv];
    negb[h * 64 + vv] = -acc;
  }
}

// ---------------------------------------------------------------------------
// out = LayerNorm(Xa + Ra) with torch-style std (ddof=1, no eps),
// elementwise gain/bias arrays (already offset to this layer). D = 512.
// ---------------------------------------------------------------------------
__global__ __launch_bounds__(256)
void add_ln_kernel(const float* __restrict__ Xa, const float* __restrict__ Ra,
                   const float* __restrict__ g, const float* __restrict__ b,
                   float* __restrict__ out) {
  __shared__ float s1[256], s2[256];
  int row = blockIdx.x, t = threadIdx.x;
  long long base = (long long)row * 512;
  float v0 = Xa[base + t]       + Ra[base + t];
  float v1 = Xa[base + 256 + t] + Ra[base + 256 + t];
  s1[t] = v0 + v1;
  s2[t] = v0 * v0 + v1 * v1;
  __syncthreads();
  for (int off = 128; off > 0; off >>= 1) {
    if (t < off) { s1[t] += s1[t + off]; s2[t] += s2[t + off]; }
    __syncthreads();
  }
  float mean = s1[0] * (1.0f / 512.0f);
  float var  = (s2[0] - 512.0f * mean * mean) * (1.0f / 511.0f);
  float inv  = rsqrtf(var);
  out[base + t]       = g[base + t]       * inv * (v0 - mean) + b[base + t];
  out[base + 256 + t] = g[base + 256 + t] * inv * (v1 - mean) + b[base + 256 + t];
}

// ---------------------------------------------------------------------------
// X[n,d] = E[ids[n], d] + PE(n, d)
// ---------------------------------------------------------------------------
__global__ __launch_bounds__(256)
void embed_pe_kernel(const int* __restrict__ ids, const float* __restrict__ E,
                     float* __restrict__ X) {
  int n = blockIdx.x;
  int id = ids[n];
  const float neg_log1e4_over_D = -9.210340371976184f / 512.0f;
  for (int d = threadIdx.x; d < 512; d += 256) {
    float e = E[(long long)id * 512 + d];
    int deven = d & ~1;
    float factor = __expf((float)deven * neg_log1e4_over_D);
    float term = (float)(n + 1) * factor;
    float p = (d & 1) ? __cosf(term) : __sinf(term);
    X[(long long)n * 512 + d] = e + p;
  }
}

// ---------------------------------------------------------------------------
// Host orchestration
// ---------------------------------------------------------------------------
extern "C" void kernel_launch(void* const* d_in, const int* in_sizes, int n_in,
                              void* d_out, int out_size, void* d_ws, size_t ws_size,
                              hipStream_t stream) {
  (void)in_sizes; (void)n_in; (void)out_size; (void)ws_size;

  constexpr int N = 2048, D = 512, DFF = 2048, H = 8, DK = 64, DV = 64, L = 6;

  const int*   ids  = (const int*)d_in[0];
  const float* E    = (const float*)d_in[1];
  const float* WQ   = (const float*)d_in[2];   // [L,H,D,DK]
  const float* WK   = (const float*)d_in[3];
  const float* WV   = (const float*)d_in[4];
  const float* WO   = (const float*)d_in[5];   // [L,H*DV,D]
  const float* ln1g = (const float*)d_in[6];   // [L,N,D]
  const float* ln1b = (const float*)d_in[7];
  const float* W1   = (const float*)d_in[8];   // [L,D,DFF]
  const float* b1   = (const float*)d_in[9];   // [L,DFF]
  const float* W2   = (const float*)d_in[10];  // [L,DFF,D]
  const float* b2   = (const float*)d_in[11];  // [L,D]
  const float* ln2g = (const float*)d_in[12];
  const float* ln2b = (const float*)d_in[13];
  float* out = (float*)d_out;

  float* ws = (float*)d_ws;
  float* X   = ws;                           // [N,D]
  float* X1  = X  + (size_t)N * D;           // [N,D]
  float* Zb  = X1 + (size_t)N * D;           // [N,D]
  float* A2  = Zb + (size_t)N * D;           // [N,H*DV]
  float* Qb  = A2 + (size_t)N * D;           // [H,N,DK]
  float* Kb  = Qb + (size_t)H * N * DK;      // [H,N,DK]
  float* Vb  = Kb + (size_t)H * N * DK;      // [H,N,DV]
  float* Fb  = Vb + (size_t)H * N * DV;      // [N,DFF]
  float* lse = Fb + (size_t)N * DFF;         // [H,N]
  float* Mm  = lse + (size_t)H * N;          // [H,DK,DV]
  float* nb  = Mm + (size_t)H * DK * DV;     // [H,DV]

  auto gemm = [&](const float* A, int lda, long long sA,
                  const float* B, int ldb, long long sB,
                  const float* bias, long long sBias,
                  float* C, int ldc, long long sC,
                  int M, int Nn, int K, int relu, int batch) {
    if (Nn % 128 == 0) {
      dim3 grid(Nn / 128, M / 128, batch);
      gemm_f16_kernel<8><<<grid, 256, 0, stream>>>(A, lda, sA, B, ldb, sB,
                                                   bias, sBias, C, ldc, sC,
                                                   M, Nn, K, relu);
    } else {
      dim3 grid(Nn / 64, M / 128, batch);
      gemm_f16_kernel<4><<<grid, 256, 0, stream>>>(A, lda, sA, B, ldb, sB,
                                                   bias, sBias, C, ldc, sC,
                                                   M, Nn, K, relu);
    }
  };

  embed_pe_kernel<<<N, 256, 0, stream>>>(ids, E, X);

  for (int l = 0; l < L; ++l) {
    const float* WQl = WQ + (size_t)l * H * D * DK;
    const float* WKl = WK + (size_t)l * H * D * DK;
    const float* WVl = WV + (size_t)l * H * D * DV;
    const float* WOl = WO + (size_t)l * H * DV * D;

    // Q/K/V projections: batched over heads (B = per-head [D,64]).
    gemm(X, D, 0, WQl, DK, (long long)D * DK, nullptr, 0,
         Qb, DK, (long long)N * DK, N, DK, D, 0, H);
    gemm(X, D, 0, WKl, DK, (long long)D * DK, nullptr, 0,
         Kb, DK, (long long)N * DK, N, DK, D, 0, H);
    gemm(X, D, 0, WVl, DV, (long long)D * DV, nullptr, 0,
         Vb, DV, (long long)N * DV, N, DV, D, 0, H);

    // Column logsumexp over the query axis (log_softmax(axis=1)).
    lse_kernel<<<dim3(N / 16, H), 32, 0, stream>>>(Qb, Kb, lse, N);

    // M[h] = K^T V / sqrt(dk);  nb[h] = -(lse . V)
    ktv_kernel<<<H, 1024, 0, stream>>>(Kb, Vb, lse, Mm, nb, N);

    // attn[h] = Q[h] @ M[h] + nb[h], written interleaved into A2 [N, H*64].
    gemm(Qb, DK, (long long)N * DK, Mm, DV, (long long)DK * DV, nb, DV,
         A2, H * DV, DV, N, DV, DK, 0, H);

    // Z = A2 @ WO
    gemm(A2, H * DV, 0, WOl, D, 0, nullptr, 0, Zb, D, 0, N, D, H * DV, 0, 1);

    // X1 = LN1(X + Z)
    add_ln_kernel<<<N, 256, 0, stream>>>(X, Zb, ln1g + (size_t)l * N * D,
                                         ln1b + (size_t)l * N * D, X1);

    // FFN
    gemm(X1, D, 0, W1 + (size_t)l * D * DFF, DFF, 0, b1 + (size_t)l * DFF, 0,
         Fb, DFF, 0, N, DFF, D, 1, 1);
    gemm(Fb, DFF, 0, W2 + (size_t)l * DFF * D, D, 0, b2 + (size_t)l * D, 0,
         Zb, D, 0, N, D, DFF, 0, 1);

    // X = LN2(X1 + R)  (last layer writes straight to d_out)
    float* dst = (l == L - 1) ? out : X;
    add_ln_kernel<<<N, 256, 0, stream>>>(X1, Zb, ln2g + (size_t)l * N * D,
                                         ln2b + (size_t)l * N * D, dst);
  }
}
